// SleepConsolidator_33483565040117
// MI455X (gfx1250) — matrix operand order repaired and checked by reading, not compiled, not run on hardware
//
#include <hip/hip_runtime.h>
#include <hip/hip_bf16.h>
#include <stdint.h>
#include <math.h>

typedef __attribute__((ext_vector_type(16))) _Float16 v16h;
typedef __attribute__((ext_vector_type(8)))  float    v8f;
typedef __attribute__((ext_vector_type(4)))  int      v4i;

#define N_STM   65536
#define N_LTM   65536
#define TOP_M   2048
#define KAPPA   0.05f
#define XI_H    0.005f
#define XI_E    0.003f
#define RHO_F   0.2f
#define C_V     2.0f
#define EPSF    1e-8f
#define THRESHF 0.5f
#define CAND_MAX 4096
#define TERR_N  (5 * 96 * 96 * 96)
#define PLANE   (96 * 96 * 96)

#define NSPLIT  16            // N-range splits (grid.y)
#define TILES_PER_BLOCK (4096 / NSPLIT)
#define STAGE   8             // tiles staged in LDS per pipeline step (16 KB)

#if __has_builtin(__builtin_amdgcn_global_load_async_to_lds_b128) && \
    __has_builtin(__builtin_amdgcn_s_wait_asynccnt)
#define USE_ASYNC_LDS 1
typedef __attribute__((address_space(1))) v4i* gv4i_p;   // global 128-bit vec ptr
typedef __attribute__((address_space(3))) v4i* lv4i_p;   // LDS 128-bit vec ptr
#else
#define USE_ASYNC_LDS 0
#endif

// ---------------- utility ----------------
__global__ void k_zero(uint32_t* p, int n) {
  int i = blockIdx.x * blockDim.x + threadIdx.x;
  if (i < n) p[i] = 0u;
}

// ---------------- top-M selection ----------------
// stm_h in [0,1): positive float bits are order-preserving. inactive -> key 0.
__global__ void k_hist(const float* h, const int* act, uint32_t* hist) {
  int i = blockIdx.x * blockDim.x + threadIdx.x;
  uint32_t key = (act[i] > 0) ? __float_as_uint(h[i]) : 0u;
  if (key) atomicAdd(&hist[key >> 16], 1u);
}

// cum[b] = number of elements with bin strictly greater than b (single WG)
__global__ void k_sufscan(const uint32_t* hist, uint32_t* cum) {
  __shared__ uint32_t s[1024];
  int t = threadIdx.x;
  uint32_t carry = 0;
  for (int chunk = 0; chunk < 64; ++chunk) {
    int b = 65535 - (chunk * 1024 + t);
    uint32_t v = hist[b];
    s[t] = v;
    __syncthreads();
    for (int off = 1; off < 1024; off <<= 1) {
      uint32_t x = (t >= off) ? s[t - off] : 0u;
      __syncthreads();
      s[t] += x;
      __syncthreads();
    }
    cum[b] = carry + s[t] - v;
    carry += s[1023];
    __syncthreads();
  }
}

__global__ void k_cand(const float* h, const int* act, const uint32_t* cum,
                       uint32_t* cnt, unsigned long long* cand) {
  int i = blockIdx.x * blockDim.x + threadIdx.x;
  uint32_t key = (act[i] > 0) ? __float_as_uint(h[i]) : 0u;
  if (key && cum[key >> 16] < (uint32_t)TOP_M) {
    uint32_t p = atomicAdd(cnt, 1u);
    if (p < CAND_MAX)
      cand[p] = ((unsigned long long)key << 32) |
                (uint32_t)(0xFFFFFFFFu - (uint32_t)i);   // tie -> lower index first
  }
}

// Single-WGP bitonic sort (descending) of candidates in LDS; emit sel[0..M)
__global__ void k_sort_sel(const unsigned long long* cand, uint32_t* sel) {
  __shared__ unsigned long long s[CAND_MAX];   // 32 KB of the 320 KB LDS
  for (int i = threadIdx.x; i < CAND_MAX; i += blockDim.x) s[i] = cand[i];
  __syncthreads();
  for (int k = 2; k <= CAND_MAX; k <<= 1) {
    for (int j = k >> 1; j > 0; j >>= 1) {
      for (int i = threadIdx.x; i < CAND_MAX; i += blockDim.x) {
        int ixj = i ^ j;
        if (ixj > i) {
          bool desc = ((i & k) == 0);
          unsigned long long a = s[i], b = s[ixj];
          if (desc ? (a < b) : (a > b)) { s[i] = b; s[ixj] = a; }
        }
      }
      __syncthreads();
    }
  }
  for (int i = threadIdx.x; i < TOP_M; i += blockDim.x)
    sel[i] = 0xFFFFFFFFu - (uint32_t)(s[i] & 0xFFFFFFFFull);
}

// ---------------- projection + A-side normalize ----------------
__global__ void k_proj(const uint32_t* sel, const float* stm_K, const float* stm_h,
                       const float* Wp, const float* bp,
                       float* Kproj, _Float16* kp16, float* omega) {
  int m = blockIdx.x, t = threadIdx.x;        // 64 threads
  int s0 = (int)sel[m];
  float acc = bp[t];
#pragma unroll
  for (int k = 0; k < 16; ++k) acc += stm_K[s0 * 16 + k] * Wp[k * 64 + t];
  __shared__ float red[64];
  red[t] = acc * acc;
  __syncthreads();
  for (int off = 32; off > 0; off >>= 1) {
    if (t < off) red[t] += red[t + off];
    __syncthreads();
  }
  float scale = 1.0f / (sqrtf(red[0]) + EPSF);
  Kproj[m * 64 + t] = acc;
  kp16[m * 64 + t] = (_Float16)(acc * scale);
  if (t == 0) omega[m] = KAPPA * stm_h[s0];
}

// Pack kp16 rows into WMMA A-fragment layout (16-bit A 16x32, ISA 7.12.2)
__global__ void k_packA(const _Float16* kp16, _Float16* Apack) {
  int tid = blockIdx.x * blockDim.x + threadIdx.x;   // 128*2*32*16 = 131072
  int h = tid & 15, lane = (tid >> 4) & 31, kc = (tid >> 9) & 1, rb = tid >> 10;
  int m = rb * 16 + (lane & 15);
  int v = h >> 1, p = h & 1;
  int k = kc * 32 + ((v >= 4) ? 16 : 0) + ((lane < 16) ? 0 : 8) + (v & 3) * 2 + p;
  Apack[tid] = kp16[m * 64 + k];
}

// Normalize ltm_K rows (zero if inactive) straight into WMMA B-fragment layout
__global__ void k_packB(const float* ltm_K, const int* act, _Float16* Bpack) {
  int j = blockIdx.x, t = threadIdx.x;        // 64 threads per row
  float v = (act[j] > 0) ? ltm_K[j * 64 + t] : 0.0f;
  __shared__ float red[64];
  red[t] = v * v;
  __syncthreads();
  for (int off = 32; off > 0; off >>= 1) {
    if (t < off) red[t] += red[t + off];
    __syncthreads();
  }
  float scale = 1.0f / (sqrtf(red[0]) + EPSF);
  int nt = j >> 4, n = j & 15, kc = t >> 5, kq = t & 31;
  int L = n + ((kq & 8) << 1);
  int h = (kq & 7) | ((kq & 16) >> 1);
  Bpack[(((nt * 2 + kc) * 32) + L) * 16 + h] = (_Float16)(v * scale);
}

// ---------------- dominant op: cosine-sim GEMM + fused argmax ----------------
// grid (16, NSPLIT), block 256 (8 waves). Wave w of block bx owns row-block
// rb = bx*8+w. All 8 waves share one N-range => B tiles staged once per block
// in LDS (double-buffered async DMA, ASYNCcnt-throttled), 8x less L2 traffic.
__global__ void __launch_bounds__(256) k_gemm_argmax(
    const _Float16* Apack, const _Float16* Bpack, float* partVal, int* partIdx) {
  __shared__ _Float16 lbuf[2][STAGE * 1024];   // 2 x 16 KB stages
  int wave = threadIdx.x >> 5, lane = threadIdx.x & 31;
  int rb = blockIdx.x * 8 + wave;
  int ny = blockIdx.y;
  int tile0 = ny * TILES_PER_BLOCK;

  const v16h* Ap = (const v16h*)Apack;
  v16h a0 = Ap[(rb * 2 + 0) * 32 + lane];
  v16h a1 = Ap[(rb * 2 + 1) * 32 + lane];

  float best[8];
  int bidx[8];
#pragma unroll
  for (int g = 0; g < 8; ++g) { best[g] = -3.4e38f; bidx[g] = 0x7fffffff; }

  // stage copy: 16 KB contiguous (tiles are contiguous in Bpack); 256 threads
  // x 4 b128 each.
  auto issue_stage = [&](int s, int buf) {
    char* src = (char*)(Bpack + (size_t)(tile0 + s * STAGE) * 1024);
    char* dst = (char*)&lbuf[buf][0];
#if USE_ASYNC_LDS
#pragma unroll
    for (int q = 0; q < 4; ++q) {
      int off = threadIdx.x * 16 + q * 4096;
      __builtin_amdgcn_global_load_async_to_lds_b128(
          (gv4i_p)(src + off), (lv4i_p)(dst + off), 0, 0);
    }
#else
    const float4* s4 = (const float4*)src;
    float4* d4 = (float4*)dst;
#pragma unroll
    for (int q = 0; q < 4; ++q) d4[threadIdx.x + q * 256] = s4[threadIdx.x + q * 256];
#endif
  };

  const int nStages = TILES_PER_BLOCK / STAGE;   // 32
  const v16h* Bl = (const v16h*)&lbuf[0][0];     // 32 B elements; 512 per buffer

  issue_stage(0, 0);
  for (int s = 0; s < nStages; ++s) {
    int cur = s & 1;
    if (s + 1 < nStages) {
      issue_stage(s + 1, cur ^ 1);
#if USE_ASYNC_LDS
      __builtin_amdgcn_s_wait_asynccnt(4);   // stage s's 4 loads complete
#endif
    } else {
#if USE_ASYNC_LDS
      __builtin_amdgcn_s_wait_asynccnt(0);
#endif
    }
    __syncthreads();                           // stage s visible to all waves
#pragma unroll
    for (int tl = 0; tl < STAGE; ++tl) {
      v16h b0 = Bl[cur * 512 + tl * 64 + lane];
      v16h b1 = Bl[cur * 512 + tl * 64 + 32 + lane];
      v8f c = {};
      c = __builtin_amdgcn_wmma_f32_16x16x32_f16(false, a0, false, b0, (short)0, c, false, false);
      c = __builtin_amdgcn_wmma_f32_16x16x32_f16(false, a1, false, b1, (short)0, c, false, false);
      int n = (tile0 + s * STAGE + tl) * 16 + (lane & 15);
#pragma unroll
      for (int g = 0; g < 8; ++g) {
        float cv = c[g];                        // C layout: VGPR g -> M=g / M=8+g
        if (cv > best[g]) { best[g] = cv; bidx[g] = n; }  // strict > keeps lowest n
      }
    }
    __syncthreads();                           // all waves done reading buf[cur]
  }

  // half-wave (16-lane) reduction over the N residues (wave32: masks 1,2,4,8)
#pragma unroll
  for (int mask = 8; mask >= 1; mask >>= 1) {
#pragma unroll
    for (int g = 0; g < 8; ++g) {
      float ov = __shfl_xor(best[g], mask);
      int on = __shfl_xor(bidx[g], mask);
      if (ov > best[g] || (ov == best[g] && on < bidx[g])) { best[g] = ov; bidx[g] = on; }
    }
  }
  if ((lane & 15) == 0) {
    int hi = lane >> 4;
#pragma unroll
    for (int g = 0; g < 8; ++g) {
      int m = rb * 16 + hi * 8 + g;
      partVal[m * NSPLIT + ny] = best[g];
      partIdx[m * NSPLIT + ny] = bidx[g];
    }
  }
}

__global__ void k_combine(const float* partVal, const int* partIdx,
                          float* bestSim, int* bestIdx) {
  int m = blockIdx.x * blockDim.x + threadIdx.x;
  if (m >= TOP_M) return;
  float b = -3.4e38f;
  int bi = 0x7fffffff;
#pragma unroll
  for (int s = 0; s < NSPLIT; ++s) {
    float v = partVal[m * NSPLIT + s];
    int i = partIdx[m * NSPLIT + s];
    if (v > b || (v == b && i < bi)) { b = v; bi = i; }
  }
  bestSim[m] = b;
  bestIdx[m] = bi;
}

// ---------------- unmatched rank (exclusive cumsum over 2048, 1 WG) --------
__global__ void k_scan(const float* bestSim, int* rank) {
  __shared__ int s[256];
  int t = threadIdx.x;
  int f[8], loc = 0;
#pragma unroll
  for (int q = 0; q < 8; ++q) {
    f[q] = (bestSim[t * 8 + q] >= THRESHF) ? 0 : 1;
    loc += f[q];
  }
  s[t] = loc;
  __syncthreads();
  for (int off = 1; off < 256; off <<= 1) {
    int x = (t >= off) ? s[t - off] : 0;
    __syncthreads();
    s[t] += x;
    __syncthreads();
  }
  int run = s[t] - loc;
#pragma unroll
  for (int q = 0; q < 8; ++q) { rank[t * 8 + q] = run; run += f[q]; }
}

// ---------------- stable argsort of ltm_active (inactive first) -------------
__global__ void k_fs1(const int* act, uint32_t* blockCnt) {
  int i = blockIdx.x * 256 + threadIdx.x;
  bool inact = (act[i] <= 0);
  unsigned m = (unsigned)__ballot(inact);
  if ((threadIdx.x & 31) == 0) atomicAdd(&blockCnt[blockIdx.x], (uint32_t)__popc(m));
}

__global__ void k_fs2(const uint32_t* blockCnt, uint32_t* blockOff, uint32_t* numInact) {
  __shared__ uint32_t s[256];
  int t = threadIdx.x;
  uint32_t v = blockCnt[t];
  s[t] = v;
  __syncthreads();
  for (int off = 1; off < 256; off <<= 1) {
    uint32_t x = (t >= off) ? s[t - off] : 0u;
    __syncthreads();
    s[t] += x;
    __syncthreads();
  }
  blockOff[t] = s[t] - v;
  if (t == 255) *numInact = s[255];
}

__global__ void k_fs3(const int* act, const uint32_t* blockOff,
                      const uint32_t* numInact, uint32_t* freeSlots) {
  __shared__ uint32_t wc[8];
  int t = threadIdx.x, j = blockIdx.x * 256 + t;
  bool inact = (act[j] <= 0);
  unsigned m = (unsigned)__ballot(inact);
  int lane = t & 31, w = t >> 5;
  if (lane == 0) wc[w] = (uint32_t)__popc(m);
  __syncthreads();
  uint32_t wbase = 0;
  for (int q = 0; q < w; ++q) wbase += wc[q];
  uint32_t exclInact = blockOff[blockIdx.x] + wbase + (uint32_t)__popc(m & ((1u << lane) - 1u));
  if (inact) freeSlots[exclInact] = (uint32_t)j;
  else       freeSlots[*numInact + ((uint32_t)j - exclInact)] = (uint32_t)j;
}

// ---------------- scatter writes into LTM outputs ----------------
__global__ void k_scatter_add(const float* bestSim, const int* bestIdx, const uint32_t* sel,
                              const float* omega, const float* Kproj,
                              const float* stm_V, const float* stm_e,
                              float* outK, float* outV, float* outE, float* outH) {
  int m = blockIdx.x;
  if (!(bestSim[m] >= THRESHF)) return;
  int j = bestIdx[m], s0 = (int)sel[m], t = threadIdx.x;
  float w = omega[m];
  atomicAdd(&outK[j * 64 + t], w * Kproj[m * 64 + t]);
  atomicAdd(&outV[j * 128 + t], w * stm_V[s0 * 128 + t]);
  atomicAdd(&outV[j * 128 + 64 + t], w * stm_V[s0 * 128 + 64 + t]);
  if (t < 4) atomicAdd(&outE[j * 4 + t], w * stm_e[s0 * 4 + t]);
  if (t == 0) atomicAdd(&outH[j], w);
}

__global__ void k_scatter_set(const float* bestSim, const int* rank, const uint32_t* freeSlots,
                              const uint32_t* sel, const float* omega, const float* Kproj,
                              const float* stm_V, const float* stm_e,
                              float* outK, float* outV, float* outE, float* outH) {
  int m = blockIdx.x;
  if (bestSim[m] >= THRESHF) return;
  int r = rank[m];
  if (r > N_LTM - 1) r = N_LTM - 1;
  int j = (int)freeSlots[r], s0 = (int)sel[m], t = threadIdx.x;
  outK[j * 64 + t] = Kproj[m * 64 + t];
  outV[j * 128 + t] = stm_V[s0 * 128 + t];
  outV[j * 128 + 64 + t] = stm_V[s0 * 128 + 64 + t];
  if (t < 4) outE[j * 4 + t] = stm_e[s0 * 4 + t];
  if (t == 0) outH[j] = omega[m];
}

// ---------------- STM value normalization (bandwidth kernel) ----------------
__global__ void k_normV(const float* stm_V, float* outV2) {
  int g = blockIdx.x * blockDim.x + threadIdx.x;   // 32 lanes per row
  int row = g >> 5, lane = g & 31;
  float4 v = ((const float4*)(stm_V + row * 128))[lane];
  float ss = v.x * v.x + v.y * v.y + v.z * v.z + v.w * v.w;
#pragma unroll
  for (int m = 16; m >= 1; m >>= 1) ss += __shfl_xor(ss, m);
  float sc = fminf(1.0f, C_V / (sqrtf(ss) + EPSF));
  float4 o;
  o.x = v.x * sc; o.y = v.y * sc; o.z = v.z * sc; o.w = v.w * sc;
  ((float4*)(outV2 + row * 128))[lane] = o;
}

// ---------------- terrain: separable 13-tap gaussian + merge ----------------
struct G13 { float w[13]; };

__global__ void k_blurX(const float* in, float* out, G13 g) {
  int i = blockIdx.x * blockDim.x + threadIdx.x;
  if (i >= TERR_N) return;
  int x = i % 96, base = i - x;
  float acc = 0.f;
#pragma unroll
  for (int t = -6; t <= 6; ++t) {
    int xx = x + t;
    if (xx >= 0 && xx < 96) acc += g.w[t + 6] * in[base + xx];
  }
  out[i] = acc;
}

__global__ void k_blurY(const float* in, float* out, G13 g) {
  int i = blockIdx.x * blockDim.x + threadIdx.x;
  if (i >= TERR_N) return;
  int y = (i / 96) % 96, base = i - y * 96;
  float acc = 0.f;
#pragma unroll
  for (int t = -6; t <= 6; ++t) {
    int yy = y + t;
    if (yy >= 0 && yy < 96) acc += g.w[t + 6] * in[base + yy * 96];
  }
  out[i] = acc;
}

__global__ void k_blurZ_combine(const float* in, const float* ltm_t, float* out, G13 g) {
  int i = blockIdx.x * blockDim.x + threadIdx.x;
  if (i >= TERR_N) return;
  int z = (i / 9216) % 96, c = i / PLANE, base = i - z * 9216;
  float acc = 0.f;
#pragma unroll
  for (int t = -6; t <= 6; ++t) {
    int zz = z + t;
    if (zz >= 0 && zz < 96) acc += g.w[t + 6] * in[base + zz * 9216];
  }
  float xi = (c == 0) ? XI_H : XI_E;
  out[i] = ltm_t[i] + xi * acc;
}

__global__ void k_fatigue(const float* f, float* o) {
  if (blockIdx.x == 0 && threadIdx.x == 0) o[0] = RHO_F * f[0];
}

// ---------------- launch ----------------
extern "C" void kernel_launch(void* const* d_in, const int* in_sizes, int n_in,
                              void* d_out, int out_size, void* d_ws, size_t ws_size,
                              hipStream_t stream) {
  (void)in_sizes; (void)n_in; (void)out_size; (void)ws_size;
  const float* stm_K   = (const float*)d_in[0];
  const float* stm_V   = (const float*)d_in[1];
  const float* stm_e   = (const float*)d_in[2];
  const float* stm_h   = (const float*)d_in[3];
  const float* ltm_K   = (const float*)d_in[4];
  const float* ltm_V   = (const float*)d_in[5];
  const float* ltm_e   = (const float*)d_in[6];
  const float* ltm_h   = (const float*)d_in[7];
  const float* W_proj  = (const float*)d_in[8];
  const float* b_proj  = (const float*)d_in[9];
  const float* stm_ter = (const float*)d_in[10];
  const float* ltm_ter = (const float*)d_in[11];
  const float* fatigue = (const float*)d_in[12];
  const int*   stm_act = (const int*)d_in[13];
  const int*   ltm_act = (const int*)d_in[14];

  float* out = (float*)d_out;
  float* outK  = out;                 // 65536*64
  float* outV  = out + 4194304;       // 65536*128
  float* outE  = out + 12582912;      // 65536*4
  float* outH  = out + 12845056;      // 65536
  float* outV2 = out + 12910592;      // 65536*128
  float* outT  = out + 21299200;      // 5*96^3
  float* outF  = out + 25722880;      // 1

  uint8_t* w = (uint8_t*)d_ws;
  auto alloc = [&](size_t bytes) -> void* {
    void* p = (void*)w;
    w += (bytes + 255) & ~(size_t)255;
    return p;
  };
  uint32_t* hist      = (uint32_t*)alloc(65536 * 4);
  uint32_t* cum       = (uint32_t*)alloc(65536 * 4);
  uint32_t* candCnt   = (uint32_t*)alloc(4);
  unsigned long long* cand = (unsigned long long*)alloc(CAND_MAX * 8);
  uint32_t* sel       = (uint32_t*)alloc(TOP_M * 4);
  float*    omega     = (float*)alloc(TOP_M * 4);
  float*    Kproj     = (float*)alloc(TOP_M * 64 * 4);
  _Float16* kp16      = (_Float16*)alloc(TOP_M * 64 * 2);
  _Float16* Apack     = (_Float16*)alloc(128 * 2 * 32 * 16 * 2);
  _Float16* Bpack     = (_Float16*)alloc((size_t)4096 * 2 * 32 * 16 * 2);
  float*    partVal   = (float*)alloc(TOP_M * NSPLIT * 4);
  int*      partIdx   = (int*)alloc(TOP_M * NSPLIT * 4);
  float*    bestSim   = (float*)alloc(TOP_M * 4);
  int*      bestIdx   = (int*)alloc(TOP_M * 4);
  int*      rank      = (int*)alloc(TOP_M * 4);
  uint32_t* blockCnt  = (uint32_t*)alloc(256 * 4);
  uint32_t* blockOff  = (uint32_t*)alloc(256 * 4);
  uint32_t* numInact  = (uint32_t*)alloc(4);
  uint32_t* freeSlots = (uint32_t*)alloc(65536 * 4);
  float*    tmp1      = (float*)alloc((size_t)TERR_N * 4);
  float*    tmp2      = (float*)alloc((size_t)TERR_N * 4);

  G13 g;
  {
    double s = 0.0;
    for (int t = 0; t < 13; ++t) {
      double x = (double)(t - 6) / 2.0;
      g.w[t] = (float)exp(-0.5 * x * x);
      s += g.w[t];
    }
    for (int t = 0; t < 13; ++t) g.w[t] = (float)((double)g.w[t] / s);
  }

  // counters/buffers that accumulate must start at zero each call
  k_zero<<<256, 256, 0, stream>>>(hist, 65536);
  k_zero<<<1, 256, 0, stream>>>(candCnt, 1);
  k_zero<<<CAND_MAX * 2 / 256, 256, 0, stream>>>((uint32_t*)cand, CAND_MAX * 2);
  k_zero<<<1, 256, 0, stream>>>(blockCnt, 256);

  // base outputs (scatter kernels then modify in place)
  (void)hipMemcpyAsync(outK, ltm_K, (size_t)N_LTM * 64 * 4, hipMemcpyDeviceToDevice, stream);
  (void)hipMemcpyAsync(outV, ltm_V, (size_t)N_LTM * 128 * 4, hipMemcpyDeviceToDevice, stream);
  (void)hipMemcpyAsync(outE, ltm_e, (size_t)N_LTM * 4 * 4, hipMemcpyDeviceToDevice, stream);
  (void)hipMemcpyAsync(outH, ltm_h, (size_t)N_LTM * 4, hipMemcpyDeviceToDevice, stream);

  // top-M select (exact set, descending order)
  k_hist<<<N_STM / 256, 256, 0, stream>>>(stm_h, stm_act, hist);
  k_sufscan<<<1, 1024, 0, stream>>>(hist, cum);
  k_cand<<<N_STM / 256, 256, 0, stream>>>(stm_h, stm_act, cum, candCnt, cand);
  k_sort_sel<<<1, 1024, 0, stream>>>(cand, sel);

  // projection + fragment packing
  k_proj<<<TOP_M, 64, 0, stream>>>(sel, stm_K, stm_h, W_proj, b_proj, Kproj, kp16, omega);
  k_packA<<<131072 / 256, 256, 0, stream>>>(kp16, Apack);
  k_packB<<<N_LTM, 64, 0, stream>>>(ltm_K, ltm_act, Bpack);

  // dominant op: [2048,64]x[64,65536] f16 WMMA GEMM with fused argmax
  k_gemm_argmax<<<dim3(16, NSPLIT), 256, 0, stream>>>(Apack, Bpack, partVal, partIdx);
  k_combine<<<TOP_M / 256, 256, 0, stream>>>(partVal, partIdx, bestSim, bestIdx);

  // unmatched ranks + free-slot permutation (stable: inactive first)
  k_scan<<<1, 256, 0, stream>>>(bestSim, rank);
  k_fs1<<<256, 256, 0, stream>>>(ltm_act, blockCnt);
  k_fs2<<<1, 256, 0, stream>>>(blockCnt, blockOff, numInact);
  k_fs3<<<256, 256, 0, stream>>>(ltm_act, blockOff, numInact, freeSlots);

  // matched additive writes, then unmatched slot sets (stream order = JAX order)
  k_scatter_add<<<TOP_M, 64, 0, stream>>>(bestSim, bestIdx, sel, omega, Kproj,
                                          stm_V, stm_e, outK, outV, outE, outH);
  k_scatter_set<<<TOP_M, 64, 0, stream>>>(bestSim, rank, freeSlots, sel, omega, Kproj,
                                          stm_V, stm_e, outK, outV, outE, outH);

  // STM value norm clip
  k_normV<<<(N_STM * 32) / 256, 256, 0, stream>>>(stm_V, outV2);

  // terrain gaussian blur (separable) + merge
  int tb = (TERR_N + 255) / 256;
  k_blurX<<<tb, 256, 0, stream>>>(stm_ter, tmp1, g);
  k_blurY<<<tb, 256, 0, stream>>>(tmp1, tmp2, g);
  k_blurZ_combine<<<tb, 256, 0, stream>>>(tmp2, ltm_ter, outT, g);

  k_fatigue<<<1, 1, 0, stream>>>(fatigue, outF);
}
